// GNNLayer_6528350290279
// MI455X (gfx1250) — matrix operand order repaired and checked by reading, not compiled, not run on hardware
//
#include <hip/hip_runtime.h>

#define DEV __device__ __forceinline__
#define AS_GLOBAL __attribute__((address_space(1)))
#define AS_LOCAL  __attribute__((address_space(3)))

typedef __attribute__((ext_vector_type(16))) __bf16 bf16x16;
typedef __attribute__((ext_vector_type(8)))  float  f32x8;
typedef int gccint4 __attribute__((vector_size(16)));  // matches builtin's V4i

static constexpr int N = 10000;
static constexpr int E = 50000;
static constexpr int D = 64;
static constexpr int H = 128;
static constexpr float SLOPE = 0.01f;

union B16x16 {
  bf16x16 v;
  unsigned u[8];
  uint4 q[2];
};
union BFBits { __bf16 h; unsigned short s; };
union PK2 { __bf16 h[2]; unsigned u; };

DEV unsigned pack_bf16(float a, float b) {
  PK2 p; p.h[0] = (__bf16)a; p.h[1] = (__bf16)b; return p.u;
}
DEV unsigned short bf16_bits(float f) { BFBits b; b.h = (__bf16)f; return b.s; }

// Guaranteed packed bf16 multiply (2 lanes per op) — avoids any f32 promotion.
DEV unsigned pkmul_bf16(unsigned a, unsigned b) {
  unsigned r;
  asm("v_pk_mul_bf16 %0, %1, %2" : "=v"(r) : "v"(a), "v"(b));
  return r;
}

DEV f32x8 wmma_bf16(bf16x16 a, bf16x16 b, f32x8 c) {
  return __builtin_amdgcn_wmma_f32_16x16x32_bf16(false, a, false, b, (short)0, c,
                                                 false, false);
}

DEV float leaky(float v) { return v >= 0.0f ? v : SLOPE * v; }

// ---------------------------------------------------------------------------
// 1. h = leaky(layernorm(x));  fp32 + bf16 copies.  one wave per node.
// ---------------------------------------------------------------------------
__global__ void k_ln(const float* __restrict__ x, const float* __restrict__ gamma,
                     const float* __restrict__ beta, float* __restrict__ hf,
                     unsigned short* __restrict__ hb) {
  int node = (blockIdx.x * blockDim.x + threadIdx.x) >> 5;
  int lane = threadIdx.x & 31;
  if (node >= N) return;
  const float* row = x + (size_t)node * D;
  float a0 = row[lane], a1 = row[lane + 32];
  float s = a0 + a1;
#pragma unroll
  for (int off = 16; off > 0; off >>= 1) s += __shfl_xor(s, off, 32);
  float mu = s * (1.0f / 64.0f);
  float d0 = a0 - mu, d1 = a1 - mu;
  float vv = d0 * d0 + d1 * d1;
#pragma unroll
  for (int off = 16; off > 0; off >>= 1) vv += __shfl_xor(vv, off, 32);
  float rstd = rsqrtf(vv * (1.0f / 64.0f) + 1e-5f);
  float h0 = leaky(d0 * rstd * gamma[lane] + beta[lane]);
  float h1 = leaky(d1 * rstd * gamma[lane + 32] + beta[lane + 32]);
  hf[(size_t)node * D + lane] = h0;
  hf[(size_t)node * D + lane + 32] = h1;
  hb[(size_t)node * D + lane] = bf16_bits(h0);
  hb[(size_t)node * D + lane + 32] = bf16_bits(h1);
}

// ---------------------------------------------------------------------------
// 2. Pre-permute w1 and w2/b2 into bf16 WMMA-B per-lane register order.
//    B layout (16-bit, 32x16): lane L -> n = L%16, kbase = (L/16)*8,
//    vgpr v holds K = (v&4?16:0)+kbase+2*(v&3) (+1) packed.
//    w1p: (s in 0..1, nt in 0..7)      w2p: (s in 0..257, nt in 0..3)
//    flatK for w2p: k*64 + d; s>=256 rows come from b2 (bias fold-in).
// ---------------------------------------------------------------------------
__global__ void k_prep(const float* __restrict__ w1, const float* __restrict__ w2,
                       const float* __restrict__ b2, unsigned* __restrict__ w1p,
                       unsigned* __restrict__ w2p) {
  int tid = blockIdx.x * blockDim.x + threadIdx.x;
  if (tid < 4096) {
    int v = tid & 7, L = (tid >> 3) & 31, nt = (tid >> 8) & 7, s = tid >> 11;
    int n = nt * 16 + (L & 15);
    int k0 = s * 32 + ((v >> 2) << 4) + ((L >> 4) << 3) + 2 * (v & 3);
    w1p[tid] = pack_bf16(w1[k0 * H + n], w1[(k0 + 1) * H + n]);
  } else {
    int idx = tid - 4096;
    if (idx >= 258 * 4 * 32 * 8) return;
    int v = idx & 7, L = (idx >> 3) & 31, nt = (idx >> 8) & 3, s = idx >> 10;
    int n = nt * 16 + (L & 15);
    int kb = (L >> 4) << 3;
    float val[2];
#pragma unroll
    for (int p = 0; p < 2; ++p) {
      int fk = s * 32 + kb + ((v >> 2) << 4) + 2 * (v & 3) + p;
      val[p] = (fk < 8192) ? w2[(size_t)(fk >> 6) * 4096 + (fk & 63) * 64 + n]
                           : b2[(fk - 8192) * 64 + n];
    }
    w2p[idx] = pack_bf16(val[0], val[1]);
  }
}

// ---------------------------------------------------------------------------
// 3. t = leaky(edge_attr @ w1 + b1) -> bf16 [E,128].  WMMA, 16 edges/wave.
// ---------------------------------------------------------------------------
__global__ void k_t(const float* __restrict__ ea, const float* __restrict__ b1,
                    const unsigned* __restrict__ w1p, unsigned short* __restrict__ tb) {
  int tile = blockIdx.x * 8 + (threadIdx.x >> 5);
  if (tile >= E / 16) return;
  int lane = threadIdx.x & 31;
  int e0 = tile * 16;
  int m = lane & 15;
  int hi8 = (lane >> 4) << 3;
  const float* rp = ea + (size_t)(e0 + m) * D;

  f32x8 C[8];
  f32x8 zero = {0.f, 0.f, 0.f, 0.f, 0.f, 0.f, 0.f, 0.f};
#pragma unroll
  for (int nt = 0; nt < 8; ++nt) C[nt] = zero;

#pragma unroll
  for (int s = 0; s < 2; ++s) {
    // A-register layout: vgpr v holds K = (v&4?16:0) + hi8 + 2*(v&3) (+1)
    B16x16 A;
    const float* bp = rp + s * 32 + hi8;
    float4 q0 = *(const float4*)(bp);
    float4 q1 = *(const float4*)(bp + 4);
    float4 q2 = *(const float4*)(bp + 16);
    float4 q3 = *(const float4*)(bp + 20);
    A.u[0] = pack_bf16(q0.x, q0.y); A.u[1] = pack_bf16(q0.z, q0.w);
    A.u[2] = pack_bf16(q1.x, q1.y); A.u[3] = pack_bf16(q1.z, q1.w);
    A.u[4] = pack_bf16(q2.x, q2.y); A.u[5] = pack_bf16(q2.z, q2.w);
    A.u[6] = pack_bf16(q3.x, q3.y); A.u[7] = pack_bf16(q3.z, q3.w);
#pragma unroll
    for (int nt = 0; nt < 8; ++nt) {
      B16x16 Bm;
      const uint4* wp = (const uint4*)(w1p + (((s * 8 + nt) * 32 + lane) << 3));
      Bm.q[0] = wp[0]; Bm.q[1] = wp[1];
      C[nt] = wmma_bf16(A.v, Bm.v, C[nt]);
    }
  }
#pragma unroll
  for (int nt = 0; nt < 8; ++nt) {
    int n = nt * 16 + m;
    float bv = b1[n];
#pragma unroll
    for (int v = 0; v < 8; ++v) {
      int M = v + hi8;  // C/D layout: row = vgpr + 8*(lane/16)
      tb[(size_t)(e0 + M) * H + n] = bf16_bits(leaky(C[nt][v] + bv));
    }
  }
}

// ---------------------------------------------------------------------------
// 4a/4b. zero accumulators; degree counts.
// ---------------------------------------------------------------------------
__global__ void k_zero(float* __restrict__ agg, float* __restrict__ cnt) {
  int i = blockIdx.x * blockDim.x + threadIdx.x;
  if (i < N * D) agg[i] = 0.f;
  if (i < N) cnt[i] = 0.f;
}

__global__ void k_cnt(const int* __restrict__ dst, float* __restrict__ cnt) {
  int e = blockIdx.x * blockDim.x + threadIdx.x;
  if (e < E) atomicAdd(&cnt[dst[e]], 1.0f);
}

// ---------------------------------------------------------------------------
// 5. The big fused kernel: msg = (t ⊗ hs) @ W2'  (K = 128*64 + 64 bias rows)
//    32 edges per wave (2 M-tiles), 4 N-tiles, scatter-add into agg.
// ---------------------------------------------------------------------------
__global__ void __launch_bounds__(256, 1) k_msg(
    const unsigned short* __restrict__ hb, const unsigned short* __restrict__ tbuf,
    const unsigned* __restrict__ w2p, const int* __restrict__ eidx,
    float* __restrict__ agg) {
  __shared__ unsigned short tlds[8][32 * H];  // 64 KB, per-wave 32x128 bf16
  const int* src = eidx;
  const int* dst = eidx + E;
  int w = threadIdx.x >> 5;
  int lane = threadIdx.x & 31;
  int tile = blockIdx.x * 8 + w;
  int e0 = tile * 32;

  // Stage this wave's t tile into LDS with XOR swizzle (conflict-free bcast).
  bool used_async = false;
  {
    int er = e0 + lane;
    unsigned* drow = (unsigned*)&tlds[w][lane * H];
    int sw = (lane & 7) << 3;  // xor in u32-index space
    if (er < E) {
      const uint4* sp = (const uint4*)(tbuf + (size_t)er * H);
#if __has_builtin(__builtin_amdgcn_global_load_async_to_lds_b128)
      // CDNA5 async copy: cache -> LDS without touching VGPRs (ASYNCcnt).
#pragma unroll
      for (int j4 = 0; j4 < 16; ++j4) {
        __builtin_amdgcn_global_load_async_to_lds_b128(
            (AS_GLOBAL gccint4*)(uint4*)(sp + j4),
            (AS_LOCAL gccint4*)&drow[(j4 * 4) ^ sw], 0, 0);
      }
      used_async = true;
#else
#pragma unroll
      for (int j4 = 0; j4 < 16; ++j4) {
        uint4 vv = sp[j4];
        *(uint4*)&drow[(j4 * 4) ^ sw] = vv;
      }
#endif
    } else {
      uint4 z = {0u, 0u, 0u, 0u};
#pragma unroll
      for (int j4 = 0; j4 < 16; ++j4) *(uint4*)&drow[(j4 * 4) ^ sw] = z;
    }
  }
  if (used_async) {
#if __has_builtin(__builtin_amdgcn_s_wait_asynccnt)
    __builtin_amdgcn_s_wait_asynccnt(0);
#else
    asm volatile("s_wait_asynccnt 0x0" ::: "memory");
#endif
  }
  __syncthreads();

  int hi8 = (lane >> 4) << 3;
  int mlo = lane & 15;

  // Gather hs[src[e]] in A-register layout, per M-tile and per 32-wide d-half.
  B16x16 hsA[2][2];
#pragma unroll
  for (int mt = 0; mt < 2; ++mt) {
    int e = e0 + mt * 16 + mlo;
    if (e >= E) e = E - 1;  // clamp; invalid rows have t==0 so A==0
    const unsigned short* hp = hb + (size_t)src[e] * D;
#pragma unroll
    for (int dh = 0; dh < 2; ++dh) {
      const unsigned short* q = hp + dh * 32 + hi8;
      hsA[mt][dh].q[0] = *(const uint4*)(q);
      hsA[mt][dh].q[1] = *(const uint4*)(q + 16);
    }
  }

  f32x8 C[2][4];
  f32x8 zero = {0.f, 0.f, 0.f, 0.f, 0.f, 0.f, 0.f, 0.f};
#pragma unroll
  for (int mt = 0; mt < 2; ++mt)
#pragma unroll
    for (int nt = 0; nt < 4; ++nt) C[mt][nt] = zero;

  for (int kk = 0; kk <= H; ++kk) {  // kk==H is the folded-b2 step (t = 1)
    unsigned tw[2];
    if (kk < H) {
#pragma unroll
      for (int mt = 0; mt < 2; ++mt) {
        int mrow = mt * 16 + mlo;
        unsigned short ts = tlds[w][mrow * H + (kk ^ ((mrow & 7) << 4))];
        tw[mt] = (unsigned)ts * 0x10001u;  // bf16 value in both halves
      }
    } else {
      tw[0] = tw[1] = 0x3F803F80u;  // bf16 1.0 x2
    }
#pragma unroll
    for (int dh = 0; dh < 2; ++dh) {
      B16x16 A0, A1;
#pragma unroll
      for (int v = 0; v < 8; ++v) {
        A0.u[v] = pkmul_bf16(hsA[0][dh].u[v], tw[0]);  // v_pk_mul_bf16
        A1.u[v] = pkmul_bf16(hsA[1][dh].u[v], tw[1]);
      }
      int step = kk * 2 + dh;  // 0..257
#pragma unroll
      for (int nt = 0; nt < 4; ++nt) {
        B16x16 Bm;
        const uint4* wp =
            (const uint4*)(w2p + (((size_t)(step * 4 + nt) * 32 + lane) << 3));
        Bm.q[0] = wp[0]; Bm.q[1] = wp[1];
        C[0][nt] = wmma_bf16(A0.v, Bm.v, C[0][nt]);
        C[1][nt] = wmma_bf16(A1.v, Bm.v, C[1][nt]);
      }
    }
  }

  // Scatter-add messages into destination accumulators.
#pragma unroll
  for (int mt = 0; mt < 2; ++mt) {
    int dn[8];
    bool ok[8];
#pragma unroll
    for (int v = 0; v < 8; ++v) {
      int e = e0 + mt * 16 + v + hi8;
      ok[v] = (e < E);
      dn[v] = ok[v] ? dst[e] : 0;
    }
#pragma unroll
    for (int nt = 0; nt < 4; ++nt) {
      int f = nt * 16 + mlo;
#pragma unroll
      for (int v = 0; v < 8; ++v) {
        if (ok[v]) atomicAdd(&agg[(size_t)dn[v] * D + f], C[mt][nt][v]);
      }
    }
  }
}

// ---------------------------------------------------------------------------
// 6. out = x + agg/max(cnt,1) + h @ root + bias     (fp32 path)
// ---------------------------------------------------------------------------
__global__ void k_final(const float* __restrict__ x, const float* __restrict__ hf,
                        const float* __restrict__ agg, const float* __restrict__ cnt,
                        const float* __restrict__ root, const float* __restrict__ bias,
                        float* __restrict__ out) {
  int i = blockIdx.x * blockDim.x + threadIdx.x;
  if (i >= N * D) return;
  int n = i >> 6, f = i & 63;
  float a = agg[i] / fmaxf(cnt[n], 1.0f);
  const float* hr = hf + (size_t)n * D;
  float r = 0.f;
#pragma unroll
  for (int d = 0; d < D; ++d) r = fmaf(hr[d], root[d * D + f], r);
  out[i] = x[i] + a + r + bias[f];
}

// ---------------------------------------------------------------------------
extern "C" void kernel_launch(void* const* d_in, const int* in_sizes, int n_in,
                              void* d_out, int out_size, void* d_ws, size_t ws_size,
                              hipStream_t stream) {
  const float* x     = (const float*)d_in[0];
  const float* ea    = (const float*)d_in[1];
  const float* gamma = (const float*)d_in[2];
  const float* beta  = (const float*)d_in[3];
  const float* w1    = (const float*)d_in[4];
  const float* b1    = (const float*)d_in[5];
  const float* w2    = (const float*)d_in[6];
  const float* b2    = (const float*)d_in[7];
  const float* root  = (const float*)d_in[8];
  const float* bias  = (const float*)d_in[9];
  const int*   eidx  = (const int*)d_in[10];
  float* out = (float*)d_out;

  char* p = (char*)d_ws;
  auto alloc = [&](size_t bytes) {
    char* r = p;
    p += (bytes + 255) & ~(size_t)255;
    return r;
  };
  float*          hf  = (float*)alloc((size_t)N * D * 4);
  unsigned short* hbb = (unsigned short*)alloc((size_t)N * D * 2);
  unsigned short* tbb = (unsigned short*)alloc((size_t)E * H * 2);
  unsigned*       w1p = (unsigned*)alloc(4096u * 4);
  unsigned*       w2p = (unsigned*)alloc(264192u * 4);
  float*          agg = (float*)alloc((size_t)N * D * 4);
  float*          cnt = (float*)alloc((size_t)N * 4);

  k_ln<<<(N + 7) / 8, 256, 0, stream>>>(x, gamma, beta, hf, hbb);
  k_prep<<<(4096 + 264192) / 256, 256, 0, stream>>>(w1, w2, b2, w1p, w2p);
  k_t<<<(E / 16 + 7) / 8, 256, 0, stream>>>(ea, b1, w1p, tbb);
  k_zero<<<(N * D + 255) / 256, 256, 0, stream>>>(agg, cnt);
  k_cnt<<<(E + 255) / 256, 256, 0, stream>>>(eidx + E, cnt);
  int tiles = (E + 31) / 32;
  k_msg<<<(tiles + 7) / 8, 256, 0, stream>>>(hbb, tbb, w2p, eidx, agg);
  k_final<<<(N * D + 255) / 256, 256, 0, stream>>>(x, hf, agg, cnt, root, bias, out);
}
